// multihead_attention_3d_4432406249853
// MI455X (gfx1250) — matrix operand order repaired and checked by reading, not compiled, not run on hardware
//
#include <hip/hip_runtime.h>
#include <hip/hip_bf16.h>

// ---------------------------------------------------------------------------
// MI455X (gfx1250) 3D multi-head attention block.
// Matmuls via v_wmma_f32_16x16x32_bf16 (wave32 WMMA); attention K/V tiles
// staged through LDS with global_load_async_to_lds_b128 (ASYNCcnt path).
// N = 4096 tokens, C = 256 channels, 8 heads x 32, 3x3x3 output conv.
// ---------------------------------------------------------------------------

typedef __bf16  bf16_t;
typedef __attribute__((ext_vector_type(16))) __bf16 v16bf;
typedef __attribute__((ext_vector_type(8)))  float  v8f;

#define NTOK   4096
#define CCH    256
#define NHEAD  8
#define HCH    32

__device__ __forceinline__ bf16_t f2bf(float f) {
  unsigned u = __builtin_bit_cast(unsigned, f);
  unsigned r = (u + 0x7FFFu + ((u >> 16) & 1u)) >> 16;   // round-nearest-even
  return __builtin_bit_cast(bf16_t, (unsigned short)r);
}
__device__ __forceinline__ bf16_t bf_zero() {
  return __builtin_bit_cast(bf16_t, (unsigned short)0);
}

__device__ __forceinline__ v8f wmma_bf16(v16bf a, v16bf b, v8f c) {
  // D = A(16x32) x B(32x16) + C, fp32 accumulate
  return __builtin_amdgcn_wmma_f32_16x16x32_bf16(false, a, false, b,
                                                 (short)0, c, false, false);
}

// A-fragment (16x32 bf16, M x K) from row-major M[ld] at (row0, k0).
__device__ __forceinline__ v16bf load_a_frag(const bf16_t* M, int row0, int k0, int ld) {
  int lane = threadIdx.x & 31;
  int m  = lane & 15;
  int hk = (lane >> 4) << 3;
  const bf16_t* p = M + (size_t)(row0 + m) * ld + k0 + hk;
  v16bf a;
#pragma unroll
  for (int e = 0; e < 8; ++e) { a[e] = p[e]; a[8 + e] = p[16 + e]; }
  return a;
}

// B-fragment (32x16 bf16, K x N) where B[k][n] = W[n0+n][k0+k], W row-major.
// Lane = column, k = 8h+e and 16+8h+e per lane half h. Works for global or LDS ptrs.
__device__ __forceinline__ v16bf load_bT_frag(const bf16_t* W, int n0, int k0, int ld) {
  int lane = threadIdx.x & 31;
  int n  = lane & 15;
  int kb = (lane >> 4) << 3;
  const bf16_t* p = W + (size_t)(n0 + n) * ld + k0 + kb;
  v16bf b;
#pragma unroll
  for (int e = 0; e < 8; ++e) { b[e] = p[e]; b[8 + e] = p[16 + e]; }
  return b;
}

__device__ __forceinline__ float half_xor_max(float v) {
#pragma unroll
  for (int off = 1; off < 16; off <<= 1) v = fmaxf(v, __shfl_xor(v, off, 32));
  return v;
}
__device__ __forceinline__ float half_xor_sum(float v) {
#pragma unroll
  for (int off = 1; off < 16; off <<= 1) v += __shfl_xor(v, off, 32);
  return v;
}

// CDNA5 async copy: LDS[lds_off + lane slot] = MEM[gaddr], 16B per lane,
// tracked with ASYNCcnt (not LOADcnt).
__device__ __forceinline__ void async_ld_b128(unsigned lds_off, const void* gaddr) {
  asm volatile("global_load_async_to_lds_b128 %0, %1, off"
               :: "v"(lds_off), "v"((unsigned long long)gaddr) : "memory");
}
__device__ __forceinline__ void wait_async0() {
  asm volatile("s_wait_asynccnt 0x0" ::: "memory");
}

// ---------------------------------------------------------------------------
// Prep kernels
// ---------------------------------------------------------------------------
__global__ void k_transpose_x(const float* __restrict__ x, bf16_t* __restrict__ xT) {
  int i = blockIdx.x * blockDim.x + threadIdx.x;      // over N*C
  if (i >= NTOK * CCH) return;
  int n = i >> 8, c = i & 255;
  xT[i] = f2bf(x[(size_t)c * NTOK + n]);
}

__global__ void k_cvt_w(const float* __restrict__ W, bf16_t* __restrict__ Wb, int n) {
  int i = blockIdx.x * blockDim.x + threadIdx.x;
  if (i < n) Wb[i] = f2bf(W[i]);
}

// Wt [CO=256][CI=256][27] -> Wtb [27][CO=256][CI=256]
__global__ void k_cvt_wt(const float* __restrict__ Wt, bf16_t* __restrict__ Wtb) {
  int i = blockIdx.x * blockDim.x + threadIdx.x;      // over 27*65536
  if (i >= 27 * CCH * CCH) return;
  int t = i >> 16;
  int rem = i & 65535;                                 // co*256 + ci
  Wtb[i] = f2bf(Wt[(size_t)rem * 27 + t]);
}

// ---------------------------------------------------------------------------
// Projection GEMM: out[n, o] = (sum_c xT[n,c] * W[o,c] + bias[o]) * scale
// ---------------------------------------------------------------------------
__global__ __launch_bounds__(128)
void k_proj(const bf16_t* __restrict__ xT, const bf16_t* __restrict__ W,
            const float* __restrict__ bias, bf16_t* __restrict__ out, float scale) {
  int wave = threadIdx.x >> 5;
  int lane = threadIdx.x & 31;
  int row0 = blockIdx.x * 64 + wave * 16;
  int col0 = blockIdx.y * 64;

  v8f acc[4] = {v8f{}, v8f{}, v8f{}, v8f{}};
  for (int k0 = 0; k0 < CCH; k0 += 32) {
    v16bf a = load_a_frag(xT, row0, k0, CCH);
#pragma unroll
    for (int t = 0; t < 4; ++t) {
      v16bf b = load_bT_frag(W, col0 + t * 16, k0, CCH);
      acc[t] = wmma_bf16(a, b, acc[t]);
    }
  }
  int rbase = row0 + ((lane >> 4) << 3);
  int cn    = lane & 15;
#pragma unroll
  for (int t = 0; t < 4; ++t) {
    int col = col0 + t * 16 + cn;
    float bv = bias[col];
#pragma unroll
    for (int r = 0; r < 8; ++r)
      out[(size_t)(rbase + r) * CCH + col] = f2bf((acc[t][r] + bv) * scale);
  }
}

// Same GEMM but stores TRANSPOSED output [256][N] (for V -> contiguous
// PV B-fragments in attention; per-lane stores are 8 contiguous bf16).
__global__ __launch_bounds__(128)
void k_proj_t(const bf16_t* __restrict__ xT, const bf16_t* __restrict__ W,
              const float* __restrict__ bias, bf16_t* __restrict__ outT) {
  int wave = threadIdx.x >> 5;
  int lane = threadIdx.x & 31;
  int row0 = blockIdx.x * 64 + wave * 16;
  int col0 = blockIdx.y * 64;

  v8f acc[4] = {v8f{}, v8f{}, v8f{}, v8f{}};
  for (int k0 = 0; k0 < CCH; k0 += 32) {
    v16bf a = load_a_frag(xT, row0, k0, CCH);
#pragma unroll
    for (int t = 0; t < 4; ++t) {
      v16bf b = load_bT_frag(W, col0 + t * 16, k0, CCH);
      acc[t] = wmma_bf16(a, b, acc[t]);
    }
  }
  int rbase = row0 + ((lane >> 4) << 3);
  int cn    = lane & 15;
#pragma unroll
  for (int t = 0; t < 4; ++t) {
    int col = col0 + t * 16 + cn;
    float bv = bias[col];
#pragma unroll
    for (int r = 0; r < 8; ++r)
      outT[(size_t)col * NTOK + rbase + r] = f2bf(acc[t][r] + bv);
  }
}

// ---------------------------------------------------------------------------
// Attention: flash-style online softmax. Wave = 32 queries of one head.
// K/V 32x32 tiles are shared by all 4 waves -> staged in double-buffered LDS
// via global_load_async_to_lds_b128 (prefetch next block during compute).
// Per 32-key block: 4 WMMAs (S), LDS repack of P, 4 WMMAs (PV).
// ---------------------------------------------------------------------------
__global__ __launch_bounds__(128)
void k_attn(const bf16_t* __restrict__ Q, const bf16_t* __restrict__ K,
            const bf16_t* __restrict__ VT, bf16_t* __restrict__ O) {
  __shared__ bf16_t ldsK[2][32][32];          // [buf][key][channel]
  __shared__ bf16_t ldsV[2][32][32];          // [buf][channel][key]
  __shared__ bf16_t ldsP[4][2][16][32];       // per-wave P repack

  int wave = threadIdx.x >> 5;
  int lane = threadIdx.x & 31;
  int h  = blockIdx.y;
  int q0 = blockIdx.x * 128 + wave * 32;
  int hc = h * HCH;

  // async staging pattern: wave w copies rows [8w, 8w+8) of each tile;
  // lane -> (row = 8w + lane/4, 16-byte segment = lane%4)
  int srow = (wave << 3) + (lane >> 2);
  int sseg = (lane & 3) << 3;                 // bf16 offset 0,8,16,24

  v16bf qa[2];
  qa[0] = load_a_frag(Q, q0,      hc, CCH);   // head dim 32 == one K step
  qa[1] = load_a_frag(Q, q0 + 16, hc, CCH);

  float m_run[2][8], l_run[2][8];
#pragma unroll
  for (int qt = 0; qt < 2; ++qt)
#pragma unroll
    for (int r = 0; r < 8; ++r) { m_run[qt][r] = -1e30f; l_run[qt][r] = 0.f; }
  v8f acc[2][2] = {{v8f{}, v8f{}}, {v8f{}, v8f{}}};

  // prime buffer 0 with the first key block
  {
    const bf16_t* gk = K  + (size_t)(0 + srow) * CCH + hc + sseg;
    const bf16_t* gv = VT + (size_t)(hc + srow) * NTOK + 0 + sseg;
    async_ld_b128((unsigned)(unsigned long long)&ldsK[0][srow][sseg], gk);
    async_ld_b128((unsigned)(unsigned long long)&ldsV[0][srow][sseg], gv);
  }

  for (int m0 = 0; m0 < NTOK; m0 += 32) {
    int buf = (m0 >> 5) & 1;
    wait_async0();            // this wave's async copies have landed
    __syncthreads();          // all waves' copies visible; prev reads done

    if (m0 + 32 < NTOK) {     // prefetch next block into the other buffer
      const bf16_t* gk = K  + (size_t)(m0 + 32 + srow) * CCH + hc + sseg;
      const bf16_t* gv = VT + (size_t)(hc + srow) * NTOK + (m0 + 32) + sseg;
      async_ld_b128((unsigned)(unsigned long long)&ldsK[buf ^ 1][srow][sseg], gk);
      async_ld_b128((unsigned)(unsigned long long)&ldsV[buf ^ 1][srow][sseg], gv);
    }

    // S = Q x K^T from LDS tile
    const bf16_t* kbase = &ldsK[buf][0][0];
    v16bf kf0 = load_bT_frag(kbase, 0,  0, 32);
    v16bf kf1 = load_bT_frag(kbase, 16, 0, 32);
    v8f z = v8f{};
    v8f s[2][2];
#pragma unroll
    for (int qt = 0; qt < 2; ++qt) {
      s[qt][0] = wmma_bf16(qa[qt], kf0, z);
      s[qt][1] = wmma_bf16(qa[qt], kf1, z);
    }

    // per-row online softmax (row 8h+r spread over the 16 lanes of half h)
#pragma unroll
    for (int qt = 0; qt < 2; ++qt) {
#pragma unroll
      for (int r = 0; r < 8; ++r) {
        float mx = half_xor_max(fmaxf(s[qt][0][r], s[qt][1][r]));
        float mnew = fmaxf(m_run[qt][r], mx);
        float alpha = __expf(m_run[qt][r] - mnew);
        m_run[qt][r] = mnew;
        float p0 = __expf(s[qt][0][r] - mnew);
        float p1 = __expf(s[qt][1][r] - mnew);
        s[qt][0][r] = p0; s[qt][1][r] = p1;
        float ls = half_xor_sum(p0 + p1);
        l_run[qt][r] = l_run[qt][r] * alpha + ls;
        acc[qt][0][r] *= alpha;
        acc[qt][1][r] *= alpha;
      }
    }

    // C-layout -> A-fragment layout via LDS (per-wave region)
#pragma unroll
    for (int qt = 0; qt < 2; ++qt)
#pragma unroll
      for (int r = 0; r < 8; ++r) {
        int rl = r + ((lane >> 4) << 3);
        ldsP[wave][qt][rl][lane & 15]        = f2bf(s[qt][0][r]);
        ldsP[wave][qt][rl][16 + (lane & 15)] = f2bf(s[qt][1][r]);
      }
    __syncthreads();
    v16bf pa[2];
    {
      int mrow = lane & 15;
      int hk   = (lane >> 4) << 3;
#pragma unroll
      for (int qt = 0; qt < 2; ++qt)
#pragma unroll
        for (int e = 0; e < 8; ++e) {
          pa[qt][e]     = ldsP[wave][qt][mrow][hk + e];
          pa[qt][8 + e] = ldsP[wave][qt][mrow][16 + hk + e];
        }
    }
    __syncthreads();

    // V fragments from LDS tile (ldsV[channel][key] -> contiguous per lane)
    const bf16_t* vbase = &ldsV[buf][0][0];
    v16bf vf0 = load_bT_frag(vbase, 0,  0, 32);
    v16bf vf1 = load_bT_frag(vbase, 16, 0, 32);
#pragma unroll
    for (int qt = 0; qt < 2; ++qt) {
      acc[qt][0] = wmma_bf16(pa[qt], vf0, acc[qt][0]);
      acc[qt][1] = wmma_bf16(pa[qt], vf1, acc[qt][1]);
    }
  }

#pragma unroll
  for (int qt = 0; qt < 2; ++qt)
#pragma unroll
    for (int r = 0; r < 8; ++r) {
      float inv = 1.f / l_run[qt][r];
      int row = q0 + qt * 16 + r + ((lane >> 4) << 3);
      O[(size_t)row * CCH + hc + (lane & 15)]      = f2bf(acc[qt][0][r] * inv);
      O[(size_t)row * CCH + hc + 16 + (lane & 15)] = f2bf(acc[qt][1][r] * inv);
    }
}

// ---------------------------------------------------------------------------
// 3x3x3 conv as implicit GEMM (27 taps x K=256) + bias + residual.
// Workgroup = 64 positions (4 waves, each one (d,h) row of 16 w) x 64 co,
// so all 4 waves share the same weight tiles through the WGP$.
// ---------------------------------------------------------------------------
__global__ __launch_bounds__(128)
void k_conv(const bf16_t* __restrict__ O, const bf16_t* __restrict__ Wtb,
            const float* __restrict__ bt, const float* __restrict__ x,
            float* __restrict__ out) {
  int wave = threadIdx.x >> 5;
  int lane = threadIdx.x & 31;
  int rowid = blockIdx.x * 4 + wave;      // 0..255 (d,h) rows
  int d  = rowid >> 4;
  int hh = rowid & 15;
  int co0 = blockIdx.y * 64;

  v8f acc[4] = {v8f{}, v8f{}, v8f{}, v8f{}};

  for (int t = 0; t < 27; ++t) {
    int dd = t / 9 - 1, dh = (t / 3) % 3 - 1, dw = t % 3 - 1;
    int nd = d + dd, nh = hh + dh;
    // prefetch next tap's weight slice into caches (global_prefetch_b8)
    if (t + 1 < 27)
      __builtin_prefetch(Wtb + (size_t)(t + 1) * CCH * CCH + (size_t)co0 * CCH, 0, 1);
    if ((unsigned)nd >= 16u || (unsigned)nh >= 16u) continue;   // wave-uniform
    const bf16_t* Wtap = Wtb + (size_t)t * CCH * CCH;
    for (int k0 = 0; k0 < CCH; k0 += 32) {
      v16bf a;
      {
        int m  = lane & 15;
        int hk = (lane >> 4) << 3;
        int nw = m + dw;                   // per-lane w-edge zero padding
        if ((unsigned)nw < 16u) {
          const bf16_t* p = O + ((size_t)((nd * 16 + nh) * 16 + nw)) * CCH + k0 + hk;
#pragma unroll
          for (int e = 0; e < 8; ++e) { a[e] = p[e]; a[8 + e] = p[16 + e]; }
        } else {
#pragma unroll
          for (int e = 0; e < 16; ++e) a[e] = bf_zero();
        }
      }
#pragma unroll
      for (int c = 0; c < 4; ++c) {
        v16bf b = load_bT_frag(Wtap, co0 + c * 16, k0, CCH);
        acc[c] = wmma_bf16(a, b, acc[c]);
      }
    }
  }

  int n_base = (d * 16 + hh) * 16;
#pragma unroll
  for (int c = 0; c < 4; ++c) {
    int co = co0 + c * 16 + (lane & 15);
    float bias = bt[co];
#pragma unroll
    for (int r = 0; r < 8; ++r) {
      int w = r + ((lane >> 4) << 3);
      size_t idx = (size_t)co * NTOK + n_base + w;
      out[idx] = x[idx] + acc[c][r] + bias;
    }
  }
}

// ---------------------------------------------------------------------------
// Host launcher
// ---------------------------------------------------------------------------
extern "C" void kernel_launch(void* const* d_in, const int* in_sizes, int n_in,
                              void* d_out, int out_size, void* d_ws, size_t ws_size,
                              hipStream_t stream) {
  const float* x  = (const float*)d_in[0];
  const float* Wq = (const float*)d_in[1];
  const float* bq = (const float*)d_in[2];
  const float* Wk = (const float*)d_in[3];
  const float* bk = (const float*)d_in[4];
  const float* Wv = (const float*)d_in[5];
  const float* bv = (const float*)d_in[6];
  const float* Wt = (const float*)d_in[7];
  const float* bt = (const float*)d_in[8];
  float* out = (float*)d_out;

  // workspace layout (bf16 elements)
  bf16_t* ws  = (bf16_t*)d_ws;
  bf16_t* xT  = ws;                       // 4096*256   [N][C]
  bf16_t* Wqb = xT  + NTOK * CCH;         // 65536
  bf16_t* Wkb = Wqb + CCH * CCH;
  bf16_t* Wvb = Wkb + CCH * CCH;
  bf16_t* Wtb = Wvb + CCH * CCH;          // 27*65536   [t][co][ci]
  bf16_t* Qb  = Wtb + 27 * CCH * CCH;     // 4096*256   [N][C]
  bf16_t* Kb  = Qb  + NTOK * CCH;         //            [N][C]
  bf16_t* VTb = Kb  + NTOK * CCH;         //            [C][N] transposed
  bf16_t* Ob  = VTb + NTOK * CCH;         //            [N][C]

  // prep
  k_transpose_x<<<(NTOK * CCH) / 256, 256, 0, stream>>>(x, xT);
  k_cvt_w<<<(CCH * CCH) / 256, 256, 0, stream>>>(Wq, Wqb, CCH * CCH);
  k_cvt_w<<<(CCH * CCH) / 256, 256, 0, stream>>>(Wk, Wkb, CCH * CCH);
  k_cvt_w<<<(CCH * CCH) / 256, 256, 0, stream>>>(Wv, Wvb, CCH * CCH);
  k_cvt_wt<<<(27 * CCH * CCH) / 256, 256, 0, stream>>>(Wt, Wtb);

  // projections (q scaled by 1/sqrt(32); V stored transposed)
  dim3 pg(NTOK / 64, CCH / 64);
  k_proj<<<pg, 128, 0, stream>>>(xT, Wqb, bq, Qb, 0.17677669529663687f);
  k_proj<<<pg, 128, 0, stream>>>(xT, Wkb, bk, Kb, 1.0f);
  k_proj_t<<<pg, 128, 0, stream>>>(xT, Wvb, bv, VTb);

  // attention: 32 query-blocks x 8 heads, wave = 32 queries
  k_attn<<<dim3(NTOK / 128, NHEAD), 128, 0, stream>>>(Qb, Kb, VTb, Ob);

  // 3x3x3 conv + bias + residual: 64 row-groups x 4 co-blocks
  k_conv<<<dim3(64, 4), 128, 0, stream>>>(Ob, Wtb, bt, x, out);
}
